// ODE_Model_87101936762955
// MI455X (gfx1250) — matrix operand order repaired and checked by reading, not compiled, not run on hardware
//
#include <hip/hip_runtime.h>

typedef __attribute__((ext_vector_type(16))) __bf16 v16bf;
typedef __attribute__((ext_vector_type(8)))  float  v8f;

namespace {
constexpr int kB = 1024, kT = 1000, kXD = 8, kZD = 8, kH = 256;
constexpr int kRowsPerWG = 16;
constexpr int kThreads   = 256;           // 8 wave32s
constexpr int kWGs       = kB / kRowsPerWG;

// LDS element counts (bf16 fragments, layout [ntile][kc][lane][i])
constexpr int W1_EL  = 16 * 1 * 32 * 16;  // folded W1': 32 x 256, single K-chunk
constexpr int W2_EL  = 16 * 8 * 32 * 16;
constexpr int W3_EL  = W2_EL;
constexpr int W4_EL  = 1 * 8 * 32 * 16;   // N padded 8->16
constexpr int INP_EL = 1 * 32 * 16;       // 16x32 A fragment (half static)
constexpr int H_EL   = 8 * 32 * 16;       // 16x256 A fragments
// fp32 state: b1,b2,b3 | b4(16) | xcur(128) | event(16) | zjump(128) | zbuf(2x128) | t(1000)
constexpr int FP_EL  = 3 * kH + 16 + kRowsPerWG * kXD + 16 + kRowsPerWG * kZD +
                       2 * kRowsPerWG * kZD + kT;
constexpr size_t SMEM_BYTES =
    (size_t)(W1_EL + W2_EL + W3_EL + W4_EL + INP_EL + 2 * H_EL) * 2 +
    (size_t)FP_EL * 4;                    // 313,376 B < 320 KB
} // namespace

// Store one scalar into the bf16 A-matrix (16xK) fragment layout (contiguous k = 16*half + i, matching B).
__device__ __forceinline__ void storeA(__bf16* frag, int kc, int M, int kk, float v) {
  int lane = M + ((kk & 16) ? 16 : 0);
  int i    = kk & 15;
  frag[(kc * 32 + lane) * 16 + i] = (__bf16)v;
}

// Swizzle a row-major fp32 [Kreal x Nreal] weight into bf16 B-matrix fragments.
__device__ __forceinline__ void swizzleW(const float* __restrict__ Wg, __bf16* frag,
                                         int Kreal, int Nreal, int Kpad, int Npad, int tid) {
  const int nkc = Kpad >> 5;
  const int tot = Kpad * Npad;
  for (int idx = tid; idx < tot; idx += kThreads) {
    int K = idx / Npad;
    int n = idx - K * Npad;
    float v = (K < Kreal && n < Nreal) ? Wg[K * Nreal + n] : 0.0f;
    int ntile = n >> 4;
    int kc    = K >> 5;
    int kk    = K & 31;
    int lane  = (n & 15) + ((kk & 16) ? 16 : 0);
    int i     = kk & 15;
    frag[((ntile * nkc + kc) * 32 + lane) * 16 + i] = (__bf16)v;
  }
}

__device__ __forceinline__ float elu_f(float x) {
  return x > 0.0f ? x : (__expf(x) - 1.0f);
}

// Async DMA one dword global -> LDS (GVS mode), tracked by ASYNCcnt.
__device__ __forceinline__ void asyncLoadZ(const float* base, int voffBytes, void* ldsPtr) {
  unsigned int ldsa = (unsigned int)(unsigned long long)ldsPtr; // low 32 = LDS offset
  asm volatile("global_load_async_to_lds_b32 %0, %1, %2 offset:0"
               :
               : "v"(ldsa), "v"(voffBytes), "s"(base)
               : "memory");
}
__device__ __forceinline__ void waitAsync0() {
  asm volatile("s_wait_asynccnt 0" ::: "memory");
}

// One dense layer for this wave's two N-tiles; wave w produces cols [32w,32w+32)
// which is exactly A-chunk kc=w of the next layer.
template <int NKC>
__device__ __forceinline__ void layerWmma(const __bf16* __restrict__ aF,
                                          const __bf16* __restrict__ wF,
                                          const float* __restrict__ bias,
                                          __bf16* __restrict__ outF,
                                          int wave, int lane) {
#pragma unroll
  for (int d = 0; d < 2; ++d) {
    const int ntile = wave * 2 + d;
    v8f acc = {};
#pragma unroll
    for (int kc = 0; kc < NKC; ++kc) {
      v16bf a = *reinterpret_cast<const v16bf*>(aF + (kc * 32 + lane) * 16);
      v16bf b = *reinterpret_cast<const v16bf*>(wF + ((ntile * NKC + kc) * 32 + lane) * 16);
      acc = __builtin_amdgcn_wmma_f32_16x16x32_bf16(false, a, false, b,
                                                    (short)0, acc, false, false);
    }
    const int   n    = lane & 15;
    const float bb   = bias[ntile * 16 + n];
    const int   mOff = (lane >= 16) ? 8 : 0;
#pragma unroll
    for (int v = 0; v < 8; ++v) {
      storeA(outF, wave, mOff + v, d * 16 + n, elu_f(acc[v] + bb));
    }
  }
}

__global__ void __launch_bounds__(kThreads, 1)
ode_persistent_kernel(const float* __restrict__ tg,  const float* __restrict__ xg,
                      const float* __restrict__ zg,  const float* __restrict__ eventg,
                      const float* __restrict__ zjumpg,
                      const float* __restrict__ w1g, const float* __restrict__ b1g,
                      const float* __restrict__ w2g, const float* __restrict__ b2g,
                      const float* __restrict__ w3g, const float* __restrict__ b3g,
                      const float* __restrict__ w4g, const float* __restrict__ b4g,
                      float* __restrict__ outg) {
  extern __shared__ __align__(32) char smemRaw[];
  __bf16* w1f  = reinterpret_cast<__bf16*>(smemRaw);
  __bf16* w2f  = w1f + W1_EL;
  __bf16* w3f  = w2f + W2_EL;
  __bf16* w4f  = w3f + W3_EL;
  __bf16* inpF = w4f + W4_EL;
  __bf16* hA   = inpF + INP_EL;
  __bf16* hB   = hA + H_EL;
  float* b1l     = reinterpret_cast<float*>(hB + H_EL);
  float* b2l     = b1l + kH;
  float* b3l     = b2l + kH;
  float* b4l     = b3l + kH;                   // 16
  float* xcur    = b4l + 16;                   // 16 x 8
  float* eventTl = xcur + kRowsPerWG * kXD;    // 16
  float* zjumpL  = eventTl + 16;               // 16 x 8
  float* zbufL   = zjumpL + kRowsPerWG * kZD;  // 2 x 16 x 8 (double buffer)
  float* tL      = zbufL + 2 * kRowsPerWG * kZD; // 1000

  const int tid  = threadIdx.x;
  const int lane = tid & 31;
  const int wave = tid >> 5;
  const int row0 = blockIdx.x * kRowsPerWG;

  // ---------------- prologue ----------------
  // Folded W1' (32 x 256): rows 0..15 = W1a - W1b ; rows 16..31 = W1b + W1c.
  for (int idx = tid; idx < 32 * kH; idx += kThreads) {
    int K = idx >> 8;
    int n = idx & (kH - 1);
    float v = (K < 16) ? (w1g[K * kH + n] - w1g[(K + 16) * kH + n])
                       : (w1g[K * kH + n] + w1g[(K + 16) * kH + n]);
    int lane2 = (n & 15) + ((K & 16) ? 16 : 0);
    w1f[(((n >> 4) * 32) + lane2) * 16 + (K & 15)] = (__bf16)v;
  }
  swizzleW(w2g, w2f, kH, kH, kH, kH, tid);
  swizzleW(w3g, w3f, kH, kH, kH, kH, tid);
  swizzleW(w4g, w4f, kH, kXD, kH, 16, tid);
  for (int i = tid; i < kH; i += kThreads) {
    b1l[i] = b1g[i]; b2l[i] = b2g[i]; b3l[i] = b3g[i];
  }
  for (int i = tid; i < kT; i += kThreads) tL[i] = tg[i];
  if (tid < 16)         b4l[tid] = (tid < kXD) ? b4g[tid] : 0.0f;
  if (tid < kRowsPerWG) eventTl[tid] = eventg[row0 + tid];
  if (tid < kRowsPerWG * kZD) zjumpL[tid] = zjumpg[row0 * kZD + tid];
  {  // static half of inp: all_initial = concat(x0, z0) -> A rows k=0..15
    int M = tid >> 4, j = tid & 15;
    float ai = (j < kXD) ? xg[(size_t)(row0 + M) * kT * kXD + j]
                         : zg[(size_t)(row0 + M) * kT * kZD + (j - kXD)];
    storeA(inpF, 0, M, j, ai);
  }
  if (tid < kRowsPerWG * kXD) {  // x state init + sol[:,0]
    int M = tid >> 3, c = tid & 7;
    float v = xg[(size_t)(row0 + M) * kT * kXD + c];
    xcur[tid] = v;
    outg[((size_t)(row0 + M) * kT + 0) * kXD + c] = v;
  }
  if (tid < kRowsPerWG * kZD) {  // async prefetch z(:, t=0) into zbuf[0]
    int M = tid >> 3, c = tid & 7;
    int voff = (int)((((size_t)(row0 + M) * kT + 0) * kZD + c) * sizeof(float));
    asyncLoadZ(zg, voff, zbufL + tid);
  }
  waitAsync0();
  __syncthreads();

  // ---------------- Euler scan: 999 sequential steps ----------------
  for (int t = 0; t < kT - 1; ++t) {
    const int buf = t & 1;
    // stage 0: dynamic half of inp (k=16..31): xtzt = [x_cur | z_eff]
    {
      const int   M    = tid >> 4;
      const int   j    = tid & 15;
      const bool  jump = (tL[t] >= eventTl[M]);
      float val;
      if (j < kXD) {
        val = xcur[M * kXD + j];
      } else {
        int jc = j - kXD;
        val = jump ? zjumpL[M * kZD + jc] : zbufL[buf * 128 + M * kZD + jc];
      }
      storeA(inpF, 0, M, 16 + j, val);
    }
    __syncthreads();
    // overlap: DMA next step's z slice into the other buffer while WMMAs run
    if (t + 1 < kT - 1 && tid < kRowsPerWG * kZD) {
      int M = tid >> 3, c = tid & 7;
      int voff = (int)((((size_t)(row0 + M) * kT + (t + 1)) * kZD + c) * sizeof(float));
      asyncLoadZ(zg, voff, zbufL + (1 - buf) * 128 + tid);
    }
    layerWmma<1>(inpF, w1f, b1l, hA, wave, lane);   // (16x32)@(32x256)
    __syncthreads();
    layerWmma<8>(hA, w2f, b2l, hB, wave, lane);     // (16x256)@(256x256)
    __syncthreads();
    layerWmma<8>(hB, w3f, b3l, hA, wave, lane);     // (16x256)@(256x256)
    __syncthreads();
    if (wave == 0) {                                // (16x256)@(256x16pad)
      v8f acc = {};
#pragma unroll
      for (int kc = 0; kc < 8; ++kc) {
        v16bf a = *reinterpret_cast<const v16bf*>(hA  + (kc * 32 + lane) * 16);
        v16bf b = *reinterpret_cast<const v16bf*>(w4f + (kc * 32 + lane) * 16);
        acc = __builtin_amdgcn_wmma_f32_16x16x32_bf16(false, a, false, b,
                                                      (short)0, acc, false, false);
      }
      const float dt   = tL[t + 1] - tL[t];
      const int   c    = lane & 15;
      const int   mOff = (lane >= 16) ? 8 : 0;
      if (c < kXD) {
#pragma unroll
        for (int v = 0; v < 8; ++v) {
          int   M  = mOff + v;
          float xn = xcur[M * kXD + c] + dt * (acc[v] + b4l[c]);
          xcur[M * kXD + c] = xn;
          outg[((size_t)(row0 + M) * kT + (t + 1)) * kXD + c] = xn;
        }
      }
    }
    waitAsync0();     // next-step z slice resident before the publishing barrier
    __syncthreads();
  }
}

extern "C" void kernel_launch(void* const* d_in, const int* in_sizes, int n_in,
                              void* d_out, int out_size, void* d_ws, size_t ws_size,
                              hipStream_t stream) {
  const float* tg = (const float*)d_in[0];
  const float* xg = (const float*)d_in[1];
  const float* zg = (const float*)d_in[2];
  const float* eg = (const float*)d_in[3];
  const float* zj = (const float*)d_in[4];
  const float* w1 = (const float*)d_in[5];
  const float* b1 = (const float*)d_in[6];
  const float* w2 = (const float*)d_in[7];
  const float* b2 = (const float*)d_in[8];
  const float* w3 = (const float*)d_in[9];
  const float* b3 = (const float*)d_in[10];
  const float* w4 = (const float*)d_in[11];
  const float* b4 = (const float*)d_in[12];
  float* out = (float*)d_out;

  (void)in_sizes; (void)n_in; (void)out_size; (void)d_ws; (void)ws_size;
  (void)hipFuncSetAttribute(reinterpret_cast<const void*>(ode_persistent_kernel),
                            hipFuncAttributeMaxDynamicSharedMemorySize,
                            (int)SMEM_BYTES);
  ode_persistent_kernel<<<kWGs, kThreads, SMEM_BYTES, stream>>>(
      tg, xg, zg, eg, zj, w1, b1, w2, b2, w3, b3, w4, b4, out);
}
